// TxtNet_59966333387000
// MI455X (gfx1250) — matrix operand order repaired and checked by reading, not compiled, not run on hardware
//
#include <hip/hip_runtime.h>
#include <hip/hip_bf16.h>
#include <math.h>

// ---------------- types for WMMA ----------------
typedef __bf16 bf16_t;
typedef __attribute__((ext_vector_type(16))) __bf16 v16bf;
typedef __attribute__((ext_vector_type(8)))  __bf16 v8bf;
typedef __attribute__((ext_vector_type(8)))  float  v8f;

#define C_IN   300
#define KPAD   320            // channels padded to multiple of 32
#define NGROUP 5
#define KTOT   (NGROUP*KPAD)  // 1600
#define LSEQ   512
#define BATCH  32
#define MROWS  (BATCH*LSEQ)   // 16384
#define HOUT   512

static __device__ __forceinline__ v16bf cat8(v8bf lo, v8bf hi) {
  return __builtin_shufflevector(lo, hi, 0,1,2,3,4,5,6,7,8,9,10,11,12,13,14,15);
}

// CDNA5 has V_TANH_F32 (transcendental VALU; co-executes with the matrix pipe).
static __device__ __forceinline__ float fast_tanh(float x) {
#if defined(__has_builtin)
#if __has_builtin(__builtin_amdgcn_tanhf)
  return __builtin_amdgcn_tanhf(x);
#else
  return tanhf(x);
#endif
#else
  return tanhf(x);
#endif
}

// ---------------- pack kernels ----------------
__global__ void __launch_bounds__(256)
pack_x_kernel(const float* __restrict__ x, bf16_t* __restrict__ X) {
  int idx = blockIdx.x * 256 + threadIdx.x;
  if (idx >= MROWS * KPAD) return;
  int row = idx / KPAD, col = idx % KPAD;
  float v = (col < C_IN) ? x[(size_t)row * C_IN + col] : 0.f;
  X[idx] = (bf16_t)v;
}

// 13 tap matrices, stored [tap][n=320][k=320], n-major / k-contiguous.
__global__ void __launch_bounds__(256)
pack_convw_kernel(const float* __restrict__ w1, const float* __restrict__ w2,
                  const float* __restrict__ w3, const float* __restrict__ w7,
                  bf16_t* __restrict__ W) {
  int idx = blockIdx.x * 256 + threadIdx.x;
  const int per = KPAD * KPAD;
  if (idx >= 13 * per) return;
  int tp = idx / per;
  int rem = idx % per;
  int o = rem / KPAD, c = rem % KPAD;   // o = out channel (N), c = in channel (K)
  float v = 0.f;
  if (o < C_IN && c < C_IN) {
    if (tp == 0)      v = w1[((size_t)o * C_IN + c) * 1 + 0];
    else if (tp < 3)  v = w2[((size_t)o * C_IN + c) * 2 + (tp - 1)];
    else if (tp < 6)  v = w3[((size_t)o * C_IN + c) * 3 + (tp - 3)];
    else              v = w7[((size_t)o * C_IN + c) * 7 + (tp - 6)];
  }
  W[idx] = (bf16_t)v;
}

// slp_w [512][1500] -> bf16 [512][1600] (per-group zero padding 300->320)
__global__ void __launch_bounds__(256)
pack_slpw_kernel(const float* __restrict__ slp_w, bf16_t* __restrict__ Wp) {
  int idx = blockIdx.x * 256 + threadIdx.x;
  if (idx >= HOUT * KTOT) return;
  int n = idx / KTOT, kk = idx % KTOT;
  int g = kk / KPAD, c = kk % KPAD;
  float v = (c < C_IN) ? slp_w[(size_t)n * (NGROUP * C_IN) + g * C_IN + c] : 0.f;
  Wp[idx] = (bf16_t)v;
}

__global__ void __launch_bounds__(256)
pack_bias_kernel(const float* __restrict__ b1, const float* __restrict__ b2,
                 const float* __restrict__ b3, const float* __restrict__ b7,
                 float* __restrict__ bp) {
  int idx = blockIdx.x * 256 + threadIdx.x;
  if (idx >= 4 * KPAD) return;
  int g = idx / KPAD, c = idx % KPAD;
  const float* s = (g == 0) ? b1 : (g == 1) ? b2 : (g == 2) ? b3 : b7;
  bp[idx] = (c < C_IN) ? s[c] : 0.f;
}

// ---------------- conv-as-GEMM (bf16 WMMA, LDS-staged B) ----------------
// Block = 128(M) x 64(N) tile: 8 waves stacked on M, sharing the B tile via LDS.
// blockIdx.x = mblock*5 + ntile64 ; blockIdx.y = conv id (0:k1, 1:k2, 2:k3, 3:k7).
__global__ void __launch_bounds__(256)
conv_wmma_kernel(const bf16_t* __restrict__ X,    // [MROWS][KPAD]
                 const bf16_t* __restrict__ W,    // [13][KPAD][KPAD]
                 const float*  __restrict__ bias, // [4][KPAD]
                 bf16_t* __restrict__ H)          // [MROWS][KTOT]
{
  const int cid = blockIdx.y;
  int tapBase, ntaps, offStart, g0, g1;
  if (cid == 0)      { tapBase = 0; ntaps = 1; offStart =  0; g0 = 0; g1 = -1; }
  else if (cid == 1) { tapBase = 1; ntaps = 2; offStart =  0; g0 = 1; g1 = -1; }
  else if (cid == 2) { tapBase = 3; ntaps = 3; offStart = -1; g0 = 2; g1 =  3; } // x5 = x3
  else               { tapBase = 6; ntaps = 7; offStart = -3; g0 = 4; g1 = -1; }

  const int mb    = blockIdx.x / 5;
  const int nt64  = blockIdx.x % 5;
  const int lane  = threadIdx.x & 31;
  const int wv    = threadIdx.x >> 5;
  const int mbase = mb * 128 + wv * 16;
  const int n0    = nt64 * 64;

  const int halfsel = lane >> 4;
  const int l15     = lane & 15;
  const int aKoff   = halfsel * 8;    // A frag: lanes 16-31 hold K {8..15,24..31}

  // double-buffered B tile: [buf][n-subtile][lane][16 bf16] = 2 x 4KB
  __shared__ __attribute__((aligned(32))) bf16_t sB[2][4][32][16];

  // staging decomposition: 256 threads x 16B = one 4KB tile per step
  const int st_s  = threadIdx.x >> 6;         // n-subtile 0..3
  const int st_ln = (threadIdx.x >> 1) & 31;  // lane slot
  const int st_hf = threadIdx.x & 1;          // 16B half
  const int st_n  = n0 + st_s * 16 + (st_ln & 15);
  const int st_kb = ((st_ln >> 4) << 3) + st_hf * 16;  // frag-internal K offset

  v8f acc0 = {}, acc1 = {}, acc2 = {}, acc3 = {};

  const int m    = mbase + l15;
  const int bidx = m >> 9;            // batch index
  const int lpos = m & (LSEQ - 1);    // position in sequence

  const int S = ntaps * (KPAD / 32);  // total (tap, kstep) steps

  // prologue: stage step 0
  {
    const bf16_t* src = W + ((size_t)tapBase * KPAD + st_n) * KPAD + st_kb;
    *(v8bf*)&sB[0][st_s][st_ln][st_hf * 8] = *(const v8bf*)src;
  }

  for (int i = 0; i < S; ++i) {
    const int pb = i & 1;
    __syncthreads();                  // staging of sB[pb] complete
    if (i + 1 < S) {                  // stage next step into the other buffer
      int tn = (i + 1) / (KPAD / 32);
      int k0n = ((i + 1) - tn * (KPAD / 32)) * 32;
      const bf16_t* src =
          W + ((size_t)(tapBase + tn) * KPAD + st_n) * KPAD + k0n + st_kb;
      *(v8bf*)&sB[pb ^ 1][st_s][st_ln][st_hf * 8] = *(const v8bf*)src;
    }
    // compute step i
    int t  = i / (KPAD / 32);
    int k0 = (i - t * (KPAD / 32)) * 32;
    int ls = lpos + offStart + t;                       // replication padding:
    ls = ls < 0 ? 0 : (ls > LSEQ - 1 ? LSEQ - 1 : ls);  // clamp within batch
    const bf16_t* arow = X + (size_t)((bidx << 9) | ls) * KPAD + aKoff;
    __builtin_prefetch(arow + k0 + 32, 0, 0);           // global_prefetch next K chunk
    v16bf A  = cat8(*(const v8bf*)(arow + k0), *(const v8bf*)(arow + k0 + 16));
    v16bf B0 = *(const v16bf*)&sB[pb][0][lane][0];
    v16bf B1 = *(const v16bf*)&sB[pb][1][lane][0];
    v16bf B2 = *(const v16bf*)&sB[pb][2][lane][0];
    v16bf B3 = *(const v16bf*)&sB[pb][3][lane][0];
    acc0 = __builtin_amdgcn_wmma_f32_16x16x32_bf16(false, A, false, B0, (short)0, acc0, false, false);
    acc1 = __builtin_amdgcn_wmma_f32_16x16x32_bf16(false, A, false, B1, (short)0, acc1, false, false);
    acc2 = __builtin_amdgcn_wmma_f32_16x16x32_bf16(false, A, false, B2, (short)0, acc2, false, false);
    acc3 = __builtin_amdgcn_wmma_f32_16x16x32_bf16(false, A, false, B3, (short)0, acc3, false, false);
  }

  // epilogue: bias + tanh, bf16 store into concat buffer (group(s) of 320 cols)
  const float bv0 = bias[cid * KPAD + n0 +  0 + l15];
  const float bv1 = bias[cid * KPAD + n0 + 16 + l15];
  const float bv2 = bias[cid * KPAD + n0 + 32 + l15];
  const float bv3 = bias[cid * KPAD + n0 + 48 + l15];
  const int rbase = mbase + halfsel * 8;   // C/D layout: VGPR r -> row r (+8 for hi lanes)
#pragma unroll
  for (int r = 0; r < 8; ++r) {
    size_t base = (size_t)(rbase + r) * KTOT;
    bf16_t o0 = (bf16_t)fast_tanh(acc0[r] + bv0);
    bf16_t o1 = (bf16_t)fast_tanh(acc1[r] + bv1);
    bf16_t o2 = (bf16_t)fast_tanh(acc2[r] + bv2);
    bf16_t o3 = (bf16_t)fast_tanh(acc3[r] + bv3);
    size_t c0 = base + (size_t)g0 * KPAD + n0 + l15;
    H[c0 +  0] = o0; H[c0 + 16] = o1; H[c0 + 32] = o2; H[c0 + 48] = o3;
    if (g1 >= 0) {
      size_t c1 = base + (size_t)g1 * KPAD + n0 + l15;
      H[c1 +  0] = o0; H[c1 + 16] = o1; H[c1 + 32] = o2; H[c1 + 48] = o3;
    }
  }
}

// ---------------- final linear (bf16 WMMA, K=1600, LDS-staged B) ----------------
// Block = 128(M) x 64(N); blockIdx.x = mblock*8 + ntile64.
__global__ void __launch_bounds__(256)
slp_wmma_kernel(const bf16_t* __restrict__ Hc,    // [MROWS][KTOT]
                const bf16_t* __restrict__ Wp,    // [512][KTOT]
                const float*  __restrict__ sb,    // [512]
                float* __restrict__ out)          // [MROWS][512]
{
  const int mb    = blockIdx.x >> 3;
  const int nt64  = blockIdx.x & 7;
  const int lane  = threadIdx.x & 31;
  const int wv    = threadIdx.x >> 5;
  const int mbase = mb * 128 + wv * 16;
  const int n0    = nt64 * 64;

  const int halfsel = lane >> 4;
  const int l15     = lane & 15;
  const int aKoff   = halfsel * 8;

  __shared__ __attribute__((aligned(32))) bf16_t sB[2][4][32][16];

  const int st_s  = threadIdx.x >> 6;
  const int st_ln = (threadIdx.x >> 1) & 31;
  const int st_hf = threadIdx.x & 1;
  const int st_n  = n0 + st_s * 16 + (st_ln & 15);
  const int st_kb = ((st_ln >> 4) << 3) + st_hf * 16;
  const bf16_t* st_base = Wp + (size_t)st_n * KTOT + st_kb;

  v8f acc0 = {}, acc1 = {}, acc2 = {}, acc3 = {};

  const bf16_t* arow = Hc + (size_t)(mbase + l15) * KTOT + aKoff;

  const int S = KTOT / 32;  // 50 K-steps

  // prologue: stage step 0
  *(v8bf*)&sB[0][st_s][st_ln][st_hf * 8] = *(const v8bf*)st_base;

  for (int i = 0; i < S; ++i) {
    const int pb = i & 1;
    __syncthreads();
    if (i + 1 < S)
      *(v8bf*)&sB[pb ^ 1][st_s][st_ln][st_hf * 8] =
          *(const v8bf*)(st_base + (i + 1) * 32);
    int k0 = i * 32;
    __builtin_prefetch(arow + k0 + 32, 0, 0);
    v16bf A  = cat8(*(const v8bf*)(arow + k0), *(const v8bf*)(arow + k0 + 16));
    v16bf B0 = *(const v16bf*)&sB[pb][0][lane][0];
    v16bf B1 = *(const v16bf*)&sB[pb][1][lane][0];
    v16bf B2 = *(const v16bf*)&sB[pb][2][lane][0];
    v16bf B3 = *(const v16bf*)&sB[pb][3][lane][0];
    acc0 = __builtin_amdgcn_wmma_f32_16x16x32_bf16(false, A, false, B0, (short)0, acc0, false, false);
    acc1 = __builtin_amdgcn_wmma_f32_16x16x32_bf16(false, A, false, B1, (short)0, acc1, false, false);
    acc2 = __builtin_amdgcn_wmma_f32_16x16x32_bf16(false, A, false, B2, (short)0, acc2, false, false);
    acc3 = __builtin_amdgcn_wmma_f32_16x16x32_bf16(false, A, false, B3, (short)0, acc3, false, false);
  }

  const float bv0 = sb[n0 +  0 + l15];
  const float bv1 = sb[n0 + 16 + l15];
  const float bv2 = sb[n0 + 32 + l15];
  const float bv3 = sb[n0 + 48 + l15];
  const int rbase = mbase + halfsel * 8;
#pragma unroll
  for (int r = 0; r < 8; ++r) {
    size_t base = (size_t)(rbase + r) * HOUT + n0 + l15;
    out[base +  0] = fast_tanh(acc0[r] + bv0);
    out[base + 16] = fast_tanh(acc1[r] + bv1);
    out[base + 32] = fast_tanh(acc2[r] + bv2);
    out[base + 48] = fast_tanh(acc3[r] + bv3);
  }
}

// ---------------- launcher ----------------
extern "C" void kernel_launch(void* const* d_in, const int* in_sizes, int n_in,
                              void* d_out, int out_size, void* d_ws, size_t ws_size,
                              hipStream_t stream) {
  (void)in_sizes; (void)n_in; (void)out_size; (void)ws_size;
  const float* x     = (const float*)d_in[0];
  const float* w1    = (const float*)d_in[1];
  const float* b1    = (const float*)d_in[2];
  const float* w2    = (const float*)d_in[3];
  const float* b2    = (const float*)d_in[4];
  const float* w3    = (const float*)d_in[5];
  const float* b3    = (const float*)d_in[6];
  const float* w7    = (const float*)d_in[7];
  const float* b7    = (const float*)d_in[8];
  const float* slp_w = (const float*)d_in[9];
  const float* slp_b = (const float*)d_in[10];
  float* out = (float*)d_out;

  char* ws = (char*)d_ws;
  const size_t szX = (size_t)MROWS * KPAD * sizeof(bf16_t);       // 10.5 MB
  const size_t szW = (size_t)13 * KPAD * KPAD * sizeof(bf16_t);   //  2.7 MB
  const size_t szS = (size_t)HOUT * KTOT * sizeof(bf16_t);        //  1.6 MB
  const size_t szB = (size_t)4 * KPAD * sizeof(float);            //  5 KB
  bf16_t* Xbf  = (bf16_t*)(ws);
  bf16_t* Wcv  = (bf16_t*)(ws + szX);
  bf16_t* Wslp = (bf16_t*)(ws + szX + szW);
  float*  Bcv  = (float*) (ws + szX + szW + szS);
  bf16_t* Hbuf = (bf16_t*)(ws + szX + szW + szS + szB);           // 52.4 MB

  // pack
  pack_x_kernel    <<<(MROWS * KPAD + 255) / 256, 256, 0, stream>>>(x, Xbf);
  pack_convw_kernel<<<(13 * KPAD * KPAD + 255) / 256, 256, 0, stream>>>(w1, w2, w3, w7, Wcv);
  pack_slpw_kernel <<<(HOUT * KTOT + 255) / 256, 256, 0, stream>>>(slp_w, Wslp);
  pack_bias_kernel <<<(4 * KPAD + 255) / 256, 256, 0, stream>>>(b1, b2, b3, b7, Bcv);

  // convs: 128 M-blocks x 5 N-tiles, 4 convs on grid.y
  conv_wmma_kernel<<<dim3((MROWS / 128) * (KPAD / 64), 4), 256, 0, stream>>>(Xbf, Wcv, Bcv, Hbuf);

  // final linear: 128 M-blocks x 8 N-tiles
  slp_wmma_kernel<<<(MROWS / 128) * (HOUT / 64), 256, 0, stream>>>(Hbuf, Wslp, slp_b, out);
}
